// Memory_33174327394644
// MI455X (gfx1250) — matrix operand order repaired and checked by reading, not compiled, not run on hardware
//
#include <hip/hip_runtime.h>
#include <hip/hip_bf16.h>

typedef __attribute__((ext_vector_type(16))) __bf16 v16bf;
typedef __attribute__((ext_vector_type(8)))  float  v8f;

#define DIMS    256
#define KCHUNKS 8      // 256 / 32
#define WAVES   8
#define TOPK    8
#define SPLITS  32
#define NEG_INF (-3.402823466e38f)

union BF16x16 { v16bf v; uint4 q[2]; uint u[8]; };

// ---------------------------------------------------------------------------
// Kernel 1: row-normalize f32 -> bf16 (cosine-similarity pre-normalization)
// one 256-thread block per row, one element per thread
// ---------------------------------------------------------------------------
__global__ void norm_rows_bf16(const float* __restrict__ in,
                               __hip_bfloat16* __restrict__ out) {
    __shared__ float red[256];
    const int row = blockIdx.x;
    const int t   = threadIdx.x;
    float x = in[(size_t)row * DIMS + t];
    red[t] = x * x;
    __syncthreads();
    for (int s = 128; s > 0; s >>= 1) {
        if (t < s) red[t] += red[t + s];
        __syncthreads();
    }
    const float inv = 1.0f / fmaxf(sqrtf(red[0]), 1e-6f);
    out[(size_t)row * DIMS + t] = __float2bfloat16(x * inv);
}

// ---------------------------------------------------------------------------
// register-resident descending top-8 insertion
// ---------------------------------------------------------------------------
__device__ __forceinline__ void topk_insert(float v, int idx,
                                            float (&bv)[TOPK], int (&bi)[TOPK]) {
    if (v <= bv[TOPK - 1]) return;
    bv[TOPK - 1] = v;
    bi[TOPK - 1] = idx;
#pragma unroll
    for (int j = TOPK - 1; j > 0; --j) {
        if (bv[j] > bv[j - 1]) {
            float tv = bv[j]; bv[j] = bv[j - 1]; bv[j - 1] = tv;
            int   ti = bi[j]; bi[j] = bi[j - 1]; bi[j - 1] = ti;
        }
    }
}

// ---------------------------------------------------------------------------
// Kernel 2: fused bf16-WMMA similarity + streaming top-8
// grid = (B/16 query tiles, SPLITS key-range splits), 256 threads = 8 waves.
// Each wave processes a 16(query) x 32(key) macro-tile per iteration with two
// independent WMMA accumulator chains so the 16 WMMAs interleave.
// ---------------------------------------------------------------------------
__global__ void sim_topk_kernel(const __hip_bfloat16* __restrict__ qn,
                                const __hip_bfloat16* __restrict__ kn,
                                int ntiles32, int tiles_per_split,
                                float* __restrict__ pval,
                                int*   __restrict__ pidx) {
    __shared__ __hip_bfloat16 qs[16 * DIMS];         //  8 KB  query tile
    __shared__ float sim[WAVES][16][32];             // 16 KB  per-wave sim tiles
    __shared__ float cval[WAVES][16][TOPK];          //  4 KB
    __shared__ int   cidx[WAVES][16][TOPK];          //  4 KB

    const int tid   = threadIdx.x;
    const int wave  = tid >> 5;
    const int lane  = tid & 31;
    const int n     = lane & 15;      // column within 16-wide tile / A row m
    const int kh    = lane >> 4;      // K half-select
    const int btile = blockIdx.x;

    // cooperative load of the 16x256 bf16 query tile into LDS
    {
        const uint* src = (const uint*)(qn + (size_t)btile * 16 * DIMS);
        uint* dst = (uint*)qs;
        for (int i = tid; i < 16 * DIMS / 2; i += 256) dst[i] = src[i];
    }
    __syncthreads();

    // build A fragments (ISA 16-bit A 16x32 layout): lane holds row m = lane&15,
    // K-pairs {2v+8*kh} for v<4 and {16+2(v-4)+8*kh} for v>=4
    BF16x16 A[KCHUNKS];
    {
        const uint* qrow = (const uint*)(qs + n * DIMS);
#pragma unroll
        for (int c = 0; c < KCHUNKS; ++c) {
            uint4 lo = *(const uint4*)(qrow + (c * 32 + kh * 8) / 2);
            uint4 hi = *(const uint4*)(qrow + (c * 32 + 16 + kh * 8) / 2);
            A[c].q[0] = lo;
            A[c].q[1] = hi;
        }
    }

    float bv[TOPK]; int bi[TOPK];
#pragma unroll
    for (int j = 0; j < TOPK; ++j) { bv[j] = NEG_INF; bi[j] = 0; }

    const int t0 = blockIdx.y * tiles_per_split;
    int t1 = t0 + tiles_per_split;
    if (t1 > ntiles32) t1 = ntiles32;

    for (int t = t0 + wave; t < t1; t += WAVES) {
        const int key_base = t * 32;

        // two adjacent 16-key tiles -> two independent accumulator chains
        const __hip_bfloat16* krow0 = kn + (size_t)(key_base + n) * DIMS;
        const __hip_bfloat16* krow1 = krow0 + (size_t)16 * DIMS;

        v8f acc0 = {0.f, 0.f, 0.f, 0.f, 0.f, 0.f, 0.f, 0.f};
        v8f acc1 = {0.f, 0.f, 0.f, 0.f, 0.f, 0.f, 0.f, 0.f};
#pragma unroll
        for (int c = 0; c < KCHUNKS; ++c) {
            BF16x16 B0, B1;
            const uint4* bp0 = (const uint4*)(krow0 + c * 32 + kh * 8);
            const uint4* bp1 = (const uint4*)(krow1 + c * 32 + kh * 8);
            B0.q[0] = bp0[0]; B0.q[1] = bp0[2];
            B1.q[0] = bp1[0]; B1.q[1] = bp1[2];
            acc0 = __builtin_amdgcn_wmma_f32_16x16x32_bf16(
                false, A[c].v, false, B0.v, (short)0, acc0, false, false);
            acc1 = __builtin_amdgcn_wmma_f32_16x16x32_bf16(
                false, A[c].v, false, B1.v, (short)0, acc1, false, false);
        }

        // near-cache prefetch of this wave's next macro-tile
        if (t + WAVES < t1)
            __builtin_prefetch(krow0 + (size_t)WAVES * 32 * DIMS, 0, 3);

        // spill both sim tiles to this wave's LDS slab (same-wave LDS in-order)
#pragma unroll
        for (int v = 0; v < 8; ++v) {
            sim[wave][v + 8 * kh][n]      = acc0[v];
            sim[wave][v + 8 * kh][16 + n] = acc1[v];
        }

        // 16 owner lanes maintain per-query-row running top-8 (float4 reads)
        if (lane < 16) {
            const float4* srow = (const float4*)&sim[wave][lane][0];
#pragma unroll
            for (int q4 = 0; q4 < 8; ++q4) {
                float4 f = srow[q4];
                topk_insert(f.x, key_base + q4 * 4 + 0, bv, bi);
                topk_insert(f.y, key_base + q4 * 4 + 1, bv, bi);
                topk_insert(f.z, key_base + q4 * 4 + 2, bv, bi);
                topk_insert(f.w, key_base + q4 * 4 + 3, bv, bi);
            }
        }
    }

    // per-wave candidates -> LDS
    if (lane < 16) {
#pragma unroll
        for (int j = 0; j < TOPK; ++j) {
            cval[wave][lane][j] = bv[j];
            cidx[wave][lane][j] = bi[j];
        }
    }
    __syncthreads();

    // wave 0 merges 8 waves x 8 candidates per row, writes partial result
    if (wave == 0 && lane < 16) {
        float fv[TOPK]; int fi[TOPK];
#pragma unroll
        for (int j = 0; j < TOPK; ++j) { fv[j] = NEG_INF; fi[j] = 0; }
        for (int w = 0; w < WAVES; ++w)
#pragma unroll
            for (int j = 0; j < TOPK; ++j)
                topk_insert(cval[w][lane][j], cidx[w][lane][j], fv, fi);

        const size_t base =
            (((size_t)btile * gridDim.y + blockIdx.y) * 16 + lane) * TOPK;
#pragma unroll
        for (int j = 0; j < TOPK; ++j) {
            pval[base + j] = fv[j];
            pidx[base + j] = fi[j];
        }
    }
}

// ---------------------------------------------------------------------------
// Kernel 3: merge SPLITS x 8 partial candidates -> final top-8 indices
// one thread per query
// ---------------------------------------------------------------------------
__global__ void merge_kernel(const float* __restrict__ pval,
                             const int*   __restrict__ pidx,
                             int splits, int B,
                             int* __restrict__ topk_idx) {
    const int b = blockIdx.x * blockDim.x + threadIdx.x;
    if (b >= B) return;
    const int btile = b >> 4;
    const int r     = b & 15;
    float fv[TOPK]; int fi[TOPK];
#pragma unroll
    for (int j = 0; j < TOPK; ++j) { fv[j] = NEG_INF; fi[j] = 0; }
    for (int s = 0; s < splits; ++s) {
        const size_t base = (((size_t)btile * splits + s) * 16 + r) * TOPK;
#pragma unroll
        for (int j = 0; j < TOPK; ++j)
            topk_insert(pval[base + j], pidx[base + j], fv, fi);
    }
#pragma unroll
    for (int j = 0; j < TOPK; ++j) topk_idx[(size_t)b * TOPK + j] = fi[j];
}

// ---------------------------------------------------------------------------
// Kernel 4: gather values[idx] -> out  (one WG per 4 KB output row, float4)
// ---------------------------------------------------------------------------
__global__ void gather_kernel(const float* __restrict__ values,
                              const int*   __restrict__ topk_idx,
                              float* __restrict__ out) {
    const int row = blockIdx.x;                 // b*TOPK + j
    const int src = topk_idx[row];
    const float4* s = (const float4*)(values + (size_t)src * 1024);
    float4*       d = (float4*)(out + (size_t)row * 1024);
    d[threadIdx.x] = s[threadIdx.x];            // 256 threads x 16 B = 4 KB
}

// ---------------------------------------------------------------------------
extern "C" void kernel_launch(void* const* d_in, const int* in_sizes, int n_in,
                              void* d_out, int out_size, void* d_ws, size_t ws_size,
                              hipStream_t stream) {
    const float* queries = (const float*)d_in[0];
    const float* keys    = (const float*)d_in[1];
    const float* values  = (const float*)d_in[2];
    // d_in[3] is top_num (device scalar) == TOPK; compile-time constant here.

    const int B = in_sizes[0] / DIMS;   // 1024
    const int M = in_sizes[1] / DIMS;   // 100000
    float* out = (float*)d_out;

    // workspace carve-up
    char* ws = (char*)d_ws;
    size_t o = 0;
    __hip_bfloat16* kn = (__hip_bfloat16*)(ws + o);
    o += (size_t)M * DIMS * sizeof(__hip_bfloat16);
    o = (o + 255) & ~(size_t)255;
    __hip_bfloat16* qn = (__hip_bfloat16*)(ws + o);
    o += (size_t)B * DIMS * sizeof(__hip_bfloat16);
    o = (o + 255) & ~(size_t)255;
    const size_t ncand = (size_t)(B / 16) * SPLITS * 16 * TOPK;
    float* pval = (float*)(ws + o); o += ncand * sizeof(float);
    o = (o + 255) & ~(size_t)255;
    int* pidx = (int*)(ws + o);     o += ncand * sizeof(int);
    o = (o + 255) & ~(size_t)255;
    int* tki = (int*)(ws + o);      o += (size_t)B * TOPK * sizeof(int);

    // 1) normalize (keys: 102 MB read + 51 MB bf16 write; bf16 keys fit in L2)
    norm_rows_bf16<<<M, 256, 0, stream>>>(keys, kn);
    norm_rows_bf16<<<B, 256, 0, stream>>>(queries, qn);

    // 2) fused WMMA similarity + streaming top-8 (32-key macro tiles)
    const int ntiles32 = M / 32;                                // 3125
    const int tps = (ntiles32 + SPLITS - 1) / SPLITS;           // 98
    sim_topk_kernel<<<dim3(B / 16, SPLITS), 256, 0, stream>>>(
        qn, kn, ntiles32, tps, pval, pidx);

    // 3) merge partials
    merge_kernel<<<(B + 255) / 256, 256, 0, stream>>>(pval, pidx, SPLITS, B, tki);

    // 4) gather values rows
    gather_kernel<<<B * TOPK, 256, 0, stream>>>(values, tki, out);
}
